// GATLayer_28106265985525
// MI455X (gfx1250) — matrix operand order repaired and checked by reading, not compiled, not run on hardware
//
#include <hip/hip_runtime.h>
#include <hip/hip_bf16.h>

// GAT layer for MI455X / gfx1250 (wave32, WMMA bf16 16x16x32 f32-acc, TDM pipeline)
// bs=8, N=2048, in_feats=256, out_feats=128

typedef __attribute__((ext_vector_type(16))) __bf16 v16bf;
typedef __attribute__((ext_vector_type(8)))  float  v8f;
typedef unsigned __attribute__((ext_vector_type(4))) u32x4;
typedef int      __attribute__((ext_vector_type(4))) i32x4;
typedef int      __attribute__((ext_vector_type(8))) i32x8;

#define GAT_ALPHA 0.2f
#define GAT_NEGINF -1e32f
#define BS    8
#define NN    2048
#define INF_  256
#define OUTF  128
#define RPB   32      // rows per block in kernel 3 (two 16-row M tiles)
#define KTILE 32

#if __has_builtin(__builtin_amdgcn_tensor_load_to_lds)
#define HAVE_TDM 1
#else
#define HAVE_TDM 0
#endif

union V16 { v16bf v; unsigned u[8]; };

// Single-instruction convert+pack: dst.lo = bf16(lo), dst.hi = bf16(hi)
__device__ inline unsigned pack_bf16(float lo, float hi) {
  unsigned r;
  asm("v_cvt_pk_bf16_f32 %0, %1, %2" : "=v"(r) : "v"(lo), "v"(hi));
  return r;
}

__device__ inline void load_A_from_P(V16& A, const unsigned short* prow, int kgrp) {
  // 16-bit A(16x32) layout: pair v at K=2v+16*kgrp (contiguous, matches B)
  const uint4* pa = (const uint4*)(prow + 16 * kgrp);
  uint4 t0 = pa[0];
  uint4 t1 = pa[1];                       // +16 bytes
  A.u[0] = t0.x; A.u[1] = t0.y; A.u[2] = t0.z; A.u[3] = t0.w;
  A.u[4] = t1.x; A.u[5] = t1.y; A.u[6] = t1.z; A.u[7] = t1.w;
}

// ---------------------------------------------------------------------------
// TDM: load a 16(rows) x 32(cols) bf16 tile (row stride NN elements) into LDS.
// D# built per CDNA5 ISA; tracked by TENSORcnt.
// ---------------------------------------------------------------------------
__device__ inline void tdm_load_tile16x32(unsigned lds_off, unsigned long long gaddr) {
#if HAVE_TDM
  u32x4 g0;
  g0.x = 1u;                                                     // count=1, user D#
  g0.y = (unsigned)__builtin_amdgcn_readfirstlane((int)lds_off); // lds_addr
  g0.z = (unsigned)__builtin_amdgcn_readfirstlane((int)(unsigned)(gaddr & 0xffffffffull));
  g0.w = (unsigned)__builtin_amdgcn_readfirstlane((int)(unsigned)(gaddr >> 32))
       | (2u << 30);                                             // type=2 ("image")
  i32x8 g1;
  g1[0] = 0x00010000;                      // wg_mask=0, data_size=1 (2B), no pad/iter
  g1[1] = (int)((NN & 0xffff) << 16);      // tensor_dim0 = 2048 (low16 in [63:48])
  g1[2] = (int)((16 & 0xffff) << 16);      // tensor_dim0 hi=0 | tensor_dim1=16
  g1[3] = (int)(KTILE << 16);              // tile_dim0 = 32 in [127:112]
  g1[4] = 16;                              // tile_dim1 = 16 in [143:128]
  g1[5] = NN;                              // tensor_dim0_stride = 2048 (low32)
  g1[6] = 0;
  g1[7] = 0;
  i32x4 z4 = {0, 0, 0, 0};
#if defined(__clang_major__) && __clang_major__ >= 23
  i32x8 z8 = {0, 0, 0, 0, 0, 0, 0, 0};
  __builtin_amdgcn_tensor_load_to_lds(g0, g1, z4, z4, z8, 0);
#else
  __builtin_amdgcn_tensor_load_to_lds(g0, g1, z4, z4, 0);
#endif
#endif
}

// ---------------------------------------------------------------------------
// Kernel 1: nodes = inputs @ W_w^T + W_b   (bf16 WMMA, f32 accumulate)
// Also writes nT[b][o][m] = bf16(nodes), K-contiguous for the aggregation GEMM.
// ---------------------------------------------------------------------------
__global__ __launch_bounds__(256) void gat_node_transform(
    const float* __restrict__ inp, const float* __restrict__ Ww,
    const float* __restrict__ Wb, float* __restrict__ nodes,
    unsigned short* __restrict__ nT)
{
  const int wave  = threadIdx.x >> 5;
  const int lane  = threadIdx.x & 31;
  const int mtile = blockIdx.x;
  const int otile = wave;
  const int m     = lane & 15;
  const int kgrp  = lane >> 4;

  const float* arow = inp + (size_t)(mtile * 16 + m) * INF_;
  const float* brow = Ww  + (size_t)(otile * 16 + m) * INF_;   // B[k][n] = Ww[n][k]

  v8f c = {};
  for (int kb = 0; kb < INF_; kb += 32) {
    V16 A, B;
#pragma unroll
    for (int v = 0; v < 8; ++v) {
      int kA = (v < 4) ? (2 * v + 8 * kgrp) : (16 + 2 * (v - 4) + 8 * kgrp);
      float2 t = *(const float2*)(arow + kb + kA);
      A.u[v] = pack_bf16(t.x, t.y);
      int kB = kA;
      float2 s = *(const float2*)(brow + kb + kB);
      B.u[v] = pack_bf16(s.x, s.y);
    }
    c = __builtin_amdgcn_wmma_f32_16x16x32_bf16(false, A.v, false, B.v,
                                                (short)0, c, false, false);
  }

  // Epilogue: tile never crosses a batch boundary (2048 % 16 == 0), so the
  // batch index and in-batch row are loop-invariant; the 8 nT elements per
  // lane are consecutive along nn -> one b128 store.
  const float bias = Wb[otile * 16 + m];
  const int g0  = mtile * 16;            // first flattened row of this tile
  const int b   = g0 >> 11;              // / NN
  const int nn0 = (g0 & (NN - 1)) + 8 * kgrp;
  const int o   = otile * 16 + m;

  float* nrow = nodes + (size_t)(g0 + 8 * kgrp) * OUTF + o;
  unsigned pk[4];
#pragma unroll
  for (int j = 0; j < 4; ++j) {
    float v0 = c[2 * j]     + bias;
    float v1 = c[2 * j + 1] + bias;
    nrow[(size_t)(2 * j) * OUTF]     = v0;
    nrow[(size_t)(2 * j + 1) * OUTF] = v1;
    pk[j] = pack_bf16(v0, v1);
  }
  uint4 st; st.x = pk[0]; st.y = pk[1]; st.z = pk[2]; st.w = pk[3];
  *(uint4*)(nT + ((size_t)(b * OUTF + o)) * NN + nn0) = st;   // 8 bf16, 16B aligned
}

// ---------------------------------------------------------------------------
// Kernel 2: f_src = nodes . a1 ; f_dst = nodes . a2
// ---------------------------------------------------------------------------
__global__ __launch_bounds__(256) void gat_attn_coeff(
    const float* __restrict__ nodes, const float* __restrict__ aw,
    float* __restrict__ fsrc, float* __restrict__ fdst)
{
  int row = blockIdx.x * blockDim.x + threadIdx.x;
  const float* nr = nodes + (size_t)row * OUTF;
  float s1 = 0.f, s2 = 0.f;
#pragma unroll 4
  for (int o = 0; o < OUTF; o += 4) {
    float4 nv = *(const float4*)(nr + o);
    float4 a1 = *(const float4*)(aw + o);
    float4 a2 = *(const float4*)(aw + OUTF + o);
    s1 += nv.x * a1.x + nv.y * a1.y + nv.z * a1.z + nv.w * a1.w;
    s2 += nv.x * a2.x + nv.y * a2.y + nv.z * a2.z + nv.w * a2.w;
  }
  fsrc[row] = s1;
  fdst[row] = s2;
}

// ---------------------------------------------------------------------------
// Kernel 3: scores -> LeakyReLU -> mask -> softmax -> attn @ nodes
// 32 rows per block. Softmax: 8 waves x 4 rows, bf16 probs into LDS P.
// GEMM: wave w owns o-cols [16w,16w+16); B tile (16x32) streamed via TDM
// into per-wave double-buffered LDS (s_wait_tensorcnt pipeline); one B load
// feeds TWO 16x16x32 WMMAs (two M tiles).
// ---------------------------------------------------------------------------
__global__ __launch_bounds__(256) void gat_softmax_aggregate(
    const int* __restrict__ adj, const float* __restrict__ fsrc,
    const float* __restrict__ fdst, const float* __restrict__ ab_p,
    const unsigned short* __restrict__ nT, float* __restrict__ out)
{
  __shared__ unsigned short P[RPB][2056];         // bf16 probs, padded stride
  __shared__ unsigned short Bt[2][8][16 * KTILE]; // TDM double buffers per wave
  __shared__ float rowsum[RPB];

  const int wave = threadIdx.x >> 5;
  const int lane = threadIdx.x & 31;
  const int b    = blockIdx.x >> 6;        // / (NN/RPB)
  const int rt   = blockIdx.x & 63;
  const float ab = ab_p[0];

  // -------- softmax phase: wave w owns rows 4w..4w+3 --------
  for (int rr = 0; rr < 4; ++rr) {
    int rl = 4 * wave + rr;
    int r  = rt * RPB + rl;
    float fs = fsrc[b * NN + r] + ab;
    const int* arow = adj + ((size_t)(b * NN + r)) * NN;

    float mx = -3.4e38f;
    for (int i = lane; i < NN; i += 32) {
      __builtin_prefetch(arow + i + 512, 0, 0);   // global_prefetch for adj stream
      float s = fs + fdst[b * NN + i];
      s = (s >= 0.f) ? s : GAT_ALPHA * s;
      s = (arow[i] > 0) ? s : GAT_NEGINF;
      mx = fmaxf(mx, s);
    }
#pragma unroll
    for (int off = 16; off > 0; off >>= 1)
      mx = fmaxf(mx, __shfl_xor(mx, off, 32));

    float sum = 0.f;
    for (int i = lane; i < NN; i += 32) {          // adj re-read hits L2
      float s = fs + fdst[b * NN + i];
      s = (s >= 0.f) ? s : GAT_ALPHA * s;
      s = (arow[i] > 0) ? s : GAT_NEGINF;
      float p = __expf(s - mx);
      sum += p;
      union { __bf16 h; unsigned short u; } cv; cv.h = (__bf16)p;
      P[rl][i] = cv.u;
    }
#pragma unroll
    for (int off = 16; off > 0; off >>= 1)
      sum += __shfl_xor(sum, off, 32);
    if (lane == 0) rowsum[rl] = sum;
  }
  __syncthreads();

  // -------- aggregation GEMM --------
  const int m    = lane & 15;
  const int kgrp = lane >> 4;
  unsigned short* buf0 = &Bt[0][wave][0];
  unsigned short* buf1 = &Bt[1][wave][0];
  const unsigned long long gbase =
      (unsigned long long)(size_t)(nT + ((size_t)(b * OUTF + wave * 16)) * NN);

#if HAVE_TDM
  tdm_load_tile16x32((unsigned)(size_t)buf0, gbase);   // prologue: first tile
#endif

  v8f c0 = {}, c1 = {};
  for (int kb = 0; kb < NN; kb += KTILE) {
    int cur = (kb >> 5) & 1;
    const unsigned short* use = cur ? buf1 : buf0;
#if HAVE_TDM
    if (kb + KTILE < NN) {
      unsigned short* nxt = cur ? buf0 : buf1;
      tdm_load_tile16x32((unsigned)(size_t)nxt, gbase + (size_t)(kb + KTILE) * 2);
      __builtin_amdgcn_s_wait_tensorcnt(1);     // current tile landed
    } else {
      __builtin_amdgcn_s_wait_tensorcnt(0);
    }
    const uint4* pb = (const uint4*)(use + m * KTILE + kgrp * 16);
    uint4 s0 = pb[0];
    uint4 s1 = pb[1];
#else
    (void)use;
    const uint4* pb = (const uint4*)(nT + ((size_t)(b * OUTF + wave * 16 + m)) * NN
                                     + kb + 16 * kgrp);
    uint4 s0 = pb[0];
    uint4 s1 = pb[1];
#endif
    V16 Bm;
    Bm.u[0] = s0.x; Bm.u[1] = s0.y; Bm.u[2] = s0.z; Bm.u[3] = s0.w;
    Bm.u[4] = s1.x; Bm.u[5] = s1.y; Bm.u[6] = s1.z; Bm.u[7] = s1.w;

    V16 A0, A1;
    load_A_from_P(A0, &P[m][kb], kgrp);
    load_A_from_P(A1, &P[16 + m][kb], kgrp);

    c0 = __builtin_amdgcn_wmma_f32_16x16x32_bf16(false, A0.v, false, Bm.v,
                                                 (short)0, c0, false, false);
    c1 = __builtin_amdgcn_wmma_f32_16x16x32_bf16(false, A1.v, false, Bm.v,
                                                 (short)0, c1, false, false);
  }

#pragma unroll
  for (int r = 0; r < 8; ++r) {
    int M0 = r + 8 * kgrp;
    int M1 = 16 + M0;
    int row0 = b * NN + rt * RPB + M0;
    int row1 = b * NN + rt * RPB + M1;
    int o = wave * 16 + m;
    out[(size_t)row0 * OUTF + o] = c0[r] * (1.0f / rowsum[M0]);
    out[(size_t)row1 * OUTF + o] = c1[r] * (1.0f / rowsum[M1]);
  }
}

// ---------------------------------------------------------------------------
extern "C" void kernel_launch(void* const* d_in, const int* in_sizes, int n_in,
                              void* d_out, int out_size, void* d_ws, size_t ws_size,
                              hipStream_t stream) {
  const float* inp = (const float*)d_in[0];
  const int*   adj = (const int*)d_in[1];
  const float* Ww  = (const float*)d_in[2];
  const float* Wb  = (const float*)d_in[3];
  const float* aw  = (const float*)d_in[4];
  const float* ab  = (const float*)d_in[5];
  float* out = (float*)d_out;

  char* ws = (char*)d_ws;
  float*          nodes = (float*)ws;                                         // 8 MB
  unsigned short* nT    = (unsigned short*)(ws + (size_t)BS * NN * OUTF * 4); // 4 MB
  float*          fsrc  = (float*)(ws + (size_t)BS * NN * OUTF * 6);
  float*          fdst  = fsrc + BS * NN;

  gat_node_transform<<<(BS * NN) / 16, 256, 0, stream>>>(inp, Ww, Wb, nodes, nT);
  gat_attn_coeff<<<(BS * NN) / 256, 256, 0, stream>>>(nodes, aw, fsrc, fdst);
  gat_softmax_aggregate<<<BS * (NN / RPB), 256, 0, stream>>>(adj, fsrc, fdst, ab, nT, out);
}
